// NGramHashEmbedding_4320737100313
// MI455X (gfx1250) — matrix operand order repaired and checked by reading, not compiled, not run on hardware
//
#include <hip/hip_runtime.h>
#include <hip/hip_bf16.h>
#include <cstdint>

// ---------------------------------------------------------------------------
// NGram hash embedding + projection, fused for MI455X (gfx1250, wave32, WMMA)
//   out[m, d] = sum_k combined[m, k] * W[d, k] + bias[d]
//   combined[m, 0:512)    = emb_2[id2[m]]   (0 rows at invalid centers)
//   combined[m, 512:1024) = emb_3[id3[m]]
//   combined[m,1024:1536) = emb_4[id4[m]]
// Kernel 1: hash ids -> d_ws.  Kernel 2: fused gather-GEMM with
// v_wmma_f32_16x16x32_bf16, double-buffered LDS, b128 LDS traffic,
// v_cvt_pk_bf16_f32 conversions.
// ---------------------------------------------------------------------------

typedef __attribute__((ext_vector_type(16))) __bf16 v16bf;
typedef __attribute__((ext_vector_type(2)))  __bf16 v2bf;
typedef __attribute__((ext_vector_type(2)))  float  v2f;
typedef __attribute__((ext_vector_type(8)))  float  v8f;

#define BATCH      8
#define SEQ        4096
#define EMBED      512
#define KTOT       1536
#define VOCAB      100000
#define M_TOTAL    (BATCH * SEQ)      // 32768

#define TM         64                 // block M tile
#define TN         128                // block N tile
#define KSTEP      32                 // WMMA K depth per step
#define NSTEPS     (KTOT / KSTEP)     // 48
#define AST        20                 // LDS A row stride (uints): 16B-aligned,
#define BST        20                 //  conflict-free (gcd(20,64)=4 -> 16 banks)
#define ABUF       (TM * AST)         // one A buffer, in uints
#define BBUF       (TN * BST)         // one B buffer, in uints

// ---------------------------------------------------------------------------
// Kernel 1: polynomial rolling hash ids for n = 2,3,4 scattered to centers.
// ids[t][m] = hash id, or -1 where the window is invalid (zero contribution).
// ---------------------------------------------------------------------------
__global__ __launch_bounds__(256) void ngram_hash_kernel(
    const int* __restrict__ bytes, int* __restrict__ ids) {
  int idx = blockIdx.x * blockDim.x + threadIdx.x;
  if (idx >= M_TOTAL) return;
  int b = idx >> 12;           // / SEQ
  int s = idx & (SEQ - 1);     // % SEQ
  const int base = b * SEQ;
#pragma unroll
  for (int t = 0; t < 3; ++t) {
    const int n = t + 2;       // 2, 3, 4
    const int c = n >> 1;      // center offset
    const int i = s - c;       // window start
    int id = -1;
    if (i >= 0 && (i + n) <= SEQ) {
      int h = 0;
#pragma unroll
      for (int j = 0; j < 4; ++j) {
        if (j < n) h = (h * 256 + (bytes[base + i + j] & 255)) % VOCAB;
      }
      id = h;
    }
    ids[t * M_TOTAL + idx] = id;
  }
}

// ---------------------------------------------------------------------------
// bf16 pair pack -> one v_cvt_pk_bf16_f32 (RNE), no manual bit surgery.
// ---------------------------------------------------------------------------
__device__ __forceinline__ unsigned pk(float x, float y) {
  v2f f;
  f.x = x;
  f.y = y;
  v2bf b = __builtin_convertvector(f, v2bf);
  return __builtin_bit_cast(unsigned, b);
}

// Per-K-step staging registers: 32 A floats/4 threads + 32 B floats/2 threads.
struct TileRegs {
  float4 a0, a1;
  float4 b0, b1, b2, b3;
};

// ---------------------------------------------------------------------------
// Kernel 2: fused gather + GEMM.
//   grid = (M_TOTAL/TM, EMBED/TN) = (512, 4), block = 256 threads = 8 waves.
//   Wave w: wm = w & 3 (M sub-tile of 16), wn = w >> 2 (N half of 64).
//   Each wave: 4 accumulators (16x16 f32) over its 16 x 64 strip.
// ---------------------------------------------------------------------------
__global__ __launch_bounds__(256) void ngram_gemm_kernel(
    const float* __restrict__ emb2, const float* __restrict__ emb3,
    const float* __restrict__ emb4, const float* __restrict__ projw,
    const float* __restrict__ bias, const int* __restrict__ ids,
    float* __restrict__ out) {
  __shared__ unsigned lds_a[2 * ABUF];   // double-buffered A tiles (10 KB)
  __shared__ unsigned lds_b[2 * BBUF];   // double-buffered B tiles (20 KB)

  const int tid   = threadIdx.x;
  const int lane  = tid & 31;
  const int wid   = tid >> 5;
  const int wm    = wid & 3;             // M sub-tile index (0..3)
  const int wn    = wid >> 2;            // N half index (0..1)
  const int l16   = lane & 15;
  const int half  = lane >> 4;           // 0: lanes 0-15, 1: lanes 16-31

  const int mBase = blockIdx.x * TM;
  const int nBase = blockIdx.y * TN;

  v8f acc[4];
#pragma unroll
  for (int t = 0; t < 4; ++t) acc[t] = (v8f)0.0f;

  // ---- hoisted gather setup: 3 ids -> 3 per-thread row base pointers ----
  const int a_row = tid >> 2;            // 0..63  (4 threads per A row)
  const int a_kc  = (tid & 3) * 8;       // float offset within 32-wide slice
  const int b_row = tid >> 1;            // 0..127 (2 threads per B row)
  const int b_kc  = (tid & 1) * 16;

  const int id0 = ids[0 * M_TOTAL + mBase + a_row];
  const int id1 = ids[1 * M_TOTAL + mBase + a_row];
  const int id2 = ids[2 * M_TOTAL + mBase + a_row];
  const float* ap0 = (id0 >= 0) ? emb2 + (size_t)id0 * EMBED + a_kc : nullptr;
  const float* ap1 = (id1 >= 0) ? emb3 + (size_t)id1 * EMBED + a_kc : nullptr;
  const float* ap2 = (id2 >= 0) ? emb4 + (size_t)id2 * EMBED + a_kc : nullptr;
  const float* wbase = projw + (size_t)(nBase + b_row) * KTOT + b_kc;

  // LDS destination pointers (per buffer)
  unsigned* a_dst0 = &lds_a[a_row * AST + (a_kc >> 1)];
  unsigned* a_dst1 = a_dst0 + ABUF;
  unsigned* b_dst0 = &lds_b[b_row * BST + (b_kc >> 1)];
  unsigned* b_dst1 = b_dst0 + BBUF;

  // LDS fragment source pointers (buffer 0; buffer 1 at +ABUF/+BBUF)
  const unsigned* afrag0 = &lds_a[(wm * 16 + l16) * AST + half * 4];
  const unsigned* bfrag0 = &lds_b[(wn * 64 + l16) * BST + half * 4];

  auto load_tile = [&](int kk) {
    TileRegs r;
    const int seg    = kk >> 4;               // table segment (16 steps each)
    const int klocal = (kk & 15) * KSTEP;     // K offset within segment
    const float* ap  = (seg == 0) ? ap0 : (seg == 1) ? ap1 : ap2;
    r.a0 = make_float4(0.f, 0.f, 0.f, 0.f);
    r.a1 = r.a0;
    if (ap != nullptr) {
      r.a0 = *(const float4*)(ap + klocal);
      r.a1 = *(const float4*)(ap + klocal + 4);
    }
    const float* wp = wbase + kk * KSTEP;
    r.b0 = *(const float4*)(wp);
    r.b1 = *(const float4*)(wp + 4);
    r.b2 = *(const float4*)(wp + 8);
    r.b3 = *(const float4*)(wp + 12);
    return r;
  };

  auto store_tile = [&](const TileRegs& r, int buf) {
    uint4 pa;
    pa.x = pk(r.a0.x, r.a0.y);
    pa.y = pk(r.a0.z, r.a0.w);
    pa.z = pk(r.a1.x, r.a1.y);
    pa.w = pk(r.a1.z, r.a1.w);
    *(uint4*)(buf ? a_dst1 : a_dst0) = pa;        // ds_store_b128
    uint4 pb0, pb1;
    pb0.x = pk(r.b0.x, r.b0.y);
    pb0.y = pk(r.b0.z, r.b0.w);
    pb0.z = pk(r.b1.x, r.b1.y);
    pb0.w = pk(r.b1.z, r.b1.w);
    pb1.x = pk(r.b2.x, r.b2.y);
    pb1.y = pk(r.b2.z, r.b2.w);
    pb1.z = pk(r.b3.x, r.b3.y);
    pb1.w = pk(r.b3.z, r.b3.w);
    unsigned* bd = buf ? b_dst1 : b_dst0;
    *(uint4*)(bd)     = pb0;                      // ds_store_b128
    *(uint4*)(bd + 4) = pb1;                      // ds_store_b128
  };

  auto compute = [&](int buf) {
    // A fragment 16x32 bf16 (ISA layout): lane L -> row L&15; VGPR j packs
    // K = (L>>4)*8 + 2j (+1); VGPR 4+j the same with K base +16.
    union { v16bf v; uint4 q[2]; } fa;
    const unsigned* ap = afrag0 + (buf ? ABUF : 0);
    fa.q[0] = *(const uint4*)(ap);                // ds_load_b128
    fa.q[1] = *(const uint4*)(ap + 8);            // ds_load_b128
#pragma unroll
    for (int t = 0; t < 4; ++t) {
      // B fragment 32x16 bf16: lane L -> column L&15; VGPR j packs
      // K = (L>>4)*8 + 2j (+1); VGPR 4+j the same with K base +16.
      union { v16bf v; uint4 q[2]; } fb;
      const unsigned* bp = bfrag0 + (buf ? BBUF : 0) + t * 16 * BST;
      fb.q[0] = *(const uint4*)(bp);              // ds_load_b128
      fb.q[1] = *(const uint4*)(bp + 8);          // ds_load_b128
      acc[t] = __builtin_amdgcn_wmma_f32_16x16x32_bf16(
          /*neg_a=*/false, fa.v, /*neg_b=*/false, fb.v,
          /*c_mod=*/(short)0, acc[t], /*reuse_a=*/false, /*reuse_b=*/false);
    }
  };

  // ---- software pipeline: load k+1 | compute k | store k+1 | barrier ----
  TileRegs r = load_tile(0);
  store_tile(r, 0);
  __syncthreads();

#pragma unroll 2
  for (int kk = 0; kk < NSTEPS - 1; ++kk) {
    TileRegs nx = load_tile(kk + 1);   // global loads overlap WMMA below
    compute(kk & 1);
    store_tile(nx, (kk + 1) & 1);      // writes the buffer not being read
    __syncthreads();
  }
  compute((NSTEPS - 1) & 1);

  // ---- epilogue: C layout (VGPR r, lane L): M = r + (L>>4)*8, N = L&15 ----
  // One base pointer per lane; all 32 stores use compile-time offsets.
  const int ncol0 = nBase + wn * 64 + l16;
  const int mrow0 = mBase + wm * 16 + half * 8;
  float* op = out + (size_t)mrow0 * EMBED + ncol0;
#pragma unroll
  for (int t = 0; t < 4; ++t) {
    const float bv = bias[ncol0 + t * 16];
#pragma unroll
    for (int rr = 0; rr < 8; ++rr) {
      op[rr * EMBED + t * 16] = acc[t][rr] + bv;
    }
  }
}

// ---------------------------------------------------------------------------
// Launch
// ---------------------------------------------------------------------------
extern "C" void kernel_launch(void* const* d_in, const int* in_sizes, int n_in,
                              void* d_out, int out_size, void* d_ws, size_t ws_size,
                              hipStream_t stream) {
  const int*   bytes = (const int*)d_in[0];   // byte_sequence [8, 4096]
  const float* emb2  = (const float*)d_in[1]; // [100000, 512]
  const float* emb3  = (const float*)d_in[2];
  const float* emb4  = (const float*)d_in[3];
  const float* projw = (const float*)d_in[4]; // [512, 1536]
  const float* projb = (const float*)d_in[5]; // [512]
  float*       out   = (float*)d_out;         // [8, 4096, 512]
  int*         ids   = (int*)d_ws;            // [3, 32768] int32 scratch

  ngram_hash_kernel<<<M_TOTAL / 256, 256, 0, stream>>>(bytes, ids);

  dim3 grid(M_TOTAL / TM, EMBED / TN);        // (512, 4)
  ngram_gemm_kernel<<<grid, 256, 0, stream>>>(emb2, emb3, emb4, projw, projb,
                                              ids, out);
}
